// GuidedAttention_24696061952714
// MI455X (gfx1250) — compile-verified, matrix operand order checked by reading, not run on hardware
//
#include <hip/hip_runtime.h>
#include <hip/hip_bf16.h>

// ---------------------------------------------------------------------------
// GuidedAttention on MI455X (gfx1250): all matrix math via v_wmma_f32_16x16x32_f16
// ---------------------------------------------------------------------------

typedef __attribute__((ext_vector_type(16))) _Float16 v16h;
typedef __attribute__((ext_vector_type(8)))  _Float16 h8;
typedef __attribute__((ext_vector_type(4)))  _Float16 h4;
typedef __attribute__((ext_vector_type(8)))  float    v8f;

#define Bsz 8
#define SQ  1024
#define D   1024
#define Hh_ 16
#define DH  64

// LDS row strides (elements) chosen so byte strides are multiples of 16
#define S_STRIDE 1028   // f32: 1028*4 = 4112 B
#define P_STRIDE 1040   // f16: 1040*2 = 2080 B
#define S_BYTES  (32 * S_STRIDE * 4)   // 131584
#define P_BYTES  (32 * P_STRIDE * 2)   // 66560
#define O_STRIDE 68
#define O_BYTES  (32 * O_STRIDE * 4)   // 8704
#define GUIDE_LDS (S_BYTES + P_BYTES)            // 198144
#define MHA_LDS   (S_BYTES + P_BYTES + O_BYTES)  // 206848

union FragU { h8 h[2]; v16h v; };

// Load a 16x32 f16 WMMA fragment (A or B role) from a K-contiguous row-major
// matrix.  p = row_base + k0 for this lane's row (row = lane&15).
// Layout (ISA 7.12.2): elems 0..7 -> K = khalf*8 + j ; elems 8..15 -> K = 16 + khalf*8 + j.
__device__ __forceinline__ v16h ldfrag(const _Float16* p, int khalf) {
  FragU u;
  u.h[0] = *reinterpret_cast<const h8*>(p + (khalf << 3));
  u.h[1] = *reinterpret_cast<const h8*>(p + 16 + (khalf << 3));
  return u.v;
}

__device__ __forceinline__ v8f wmma_f16(v16h a, v16h b, v8f c) {
  return __builtin_amdgcn_wmma_f32_16x16x32_f16(false, a, false, b, (short)0, c,
                                                false, false);
}

__device__ __forceinline__ v8f vzero8() {
  v8f z = {0.f, 0.f, 0.f, 0.f, 0.f, 0.f, 0.f, 0.f};
  return z;
}

// ---------------------------------------------------------------------------
// f32 -> f16 conversion (n multiple of 4)
// ---------------------------------------------------------------------------
__global__ __launch_bounds__(256, 1) void f32_to_f16_kernel(
    const float* __restrict__ in, _Float16* __restrict__ out, int n) {
  int i = (blockIdx.x * blockDim.x + threadIdx.x) << 2;
  if (i < n) {
    float4 v = *reinterpret_cast<const float4*>(in + i);
    h4 o;
    o.x = (_Float16)v.x; o.y = (_Float16)v.y;
    o.z = (_Float16)v.z; o.w = (_Float16)v.w;
    *reinterpret_cast<h4*>(out + i) = o;
  }
}

// ---------------------------------------------------------------------------
// C[M,N] = act(A[M,K] @ W[N,K]^T + bias[N]);  f16 in, f16 out, f32 accum.
// One wave computes a 64x64 tile. block = (32,8) -> 8 waves/block.
// launch_bounds(256,1): 2 waves/SIMD32, full VGPR budget -> no accum spills.
// ---------------------------------------------------------------------------
__global__ __launch_bounds__(256, 1) void gemm_bias_kernel(
    const _Float16* __restrict__ A, const _Float16* __restrict__ W,
    const float* __restrict__ bias, _Float16* __restrict__ C,
    int M, int N, int K, int relu) {
  const int lane  = threadIdx.x;
  const int khalf = lane >> 4;
  const int lr    = lane & 15;
  const int wid   = blockIdx.x * blockDim.y + threadIdx.y;
  const int wpn   = N >> 6;                // 64-col tiles across N
  const int mt    = wid / wpn, nt = wid % wpn;
  const int m0    = mt << 6,  n0 = nt << 6;
  if (m0 >= M) return;

  v8f acc[4][4];
#pragma unroll
  for (int i = 0; i < 4; ++i)
#pragma unroll
    for (int j = 0; j < 4; ++j) acc[i][j] = vzero8();

  const _Float16* Ap = A + (size_t)(m0 + lr) * K;
  const _Float16* Wp = W + (size_t)(n0 + lr) * K;

  for (int k0 = 0; k0 < K; k0 += 32) {
    v16h a[4];
#pragma unroll
    for (int i = 0; i < 4; ++i)
      a[i] = ldfrag(Ap + (size_t)(i << 4) * K + k0, khalf);
#pragma unroll
    for (int j = 0; j < 4; ++j) {
      v16h b = ldfrag(Wp + (size_t)(j << 4) * K + k0, khalf);
#pragma unroll
      for (int i = 0; i < 4; ++i) acc[i][j] = wmma_f16(a[i], b, acc[i][j]);
    }
  }

  const int rbase = m0 + (khalf << 3);
#pragma unroll
  for (int j = 0; j < 4; ++j) {
    const int col = n0 + (j << 4) + lr;
    const float bv = bias[col];
#pragma unroll
    for (int i = 0; i < 4; ++i) {
#pragma unroll
      for (int e = 0; e < 8; ++e) {
        float v = acc[i][j][e] + bv;
        if (relu) v = fmaxf(v, 0.f);
        C[(size_t)(rbase + (i << 4) + e) * N + col] = (_Float16)v;
      }
    }
  }
}

// ---------------------------------------------------------------------------
// Per-batch transpose: Vt[b][n][k] = V[b][k][n]   (1024x1024 f16)
// ---------------------------------------------------------------------------
__global__ __launch_bounds__(512, 1) void transpose_kernel(
    const _Float16* __restrict__ V, _Float16* __restrict__ Vt) {
  __shared__ _Float16 t[64][72];
  const size_t base = (size_t)blockIdx.z << 20;
  const int nx = blockIdx.x << 6, ky = blockIdx.y << 6;
  const int tx = threadIdx.x, ty = threadIdx.y;
#pragma unroll
  for (int i = 0; i < 8; ++i)
    t[ty + (i << 3)][tx] = V[base + (size_t)(ky + ty + (i << 3)) * 1024 + nx + tx];
  __syncthreads();
#pragma unroll
  for (int i = 0; i < 8; ++i)
    Vt[base + (size_t)(nx + ty + (i << 3)) * 1024 + ky + tx] = t[tx][ty + (i << 3)];
}

// ---------------------------------------------------------------------------
// row softmax over 32 rows x 1024 cols held in LDS; wave w handles rows 4w..4w+3
// ---------------------------------------------------------------------------
__device__ __forceinline__ void softmax_rows(float* S, _Float16* P, int w, int lane) {
  for (int rr = 0; rr < 4; ++rr) {
    float*    Srow = S + (size_t)(w * 4 + rr) * S_STRIDE;
    _Float16* Prow = P + (size_t)(w * 4 + rr) * P_STRIDE;
    float m = -3.0e38f;
#pragma unroll
    for (int i = 0; i < 32; ++i) m = fmaxf(m, Srow[lane + (i << 5)]);
#pragma unroll
    for (int off = 16; off > 0; off >>= 1) m = fmaxf(m, __shfl_xor(m, off, 32));
    float s = 0.f;
#pragma unroll
    for (int i = 0; i < 32; ++i) {
      float e = __expf(Srow[lane + (i << 5)] - m);
      Srow[lane + (i << 5)] = e;
      s += e;
    }
#pragma unroll
    for (int off = 16; off > 0; off >>= 1) s += __shfl_xor(s, off, 32);
    const float inv = 1.f / s;
#pragma unroll
    for (int i = 0; i < 32; ++i)
      Prow[lane + (i << 5)] = (_Float16)(Srow[lane + (i << 5)] * inv);
  }
}

// ---------------------------------------------------------------------------
// Guide attention (single head over full D): G = softmax(Q K^T / 32) @ V
// One block (8 waves) per (batch, 32-query tile). Wave w owns key/col slab
// [128w,128w+128).  S: LDS f32 [32][1028]; P: LDS f16 [32][1040].
// ---------------------------------------------------------------------------
__global__ __launch_bounds__(256, 1) void guide_attn_kernel(
    const _Float16* __restrict__ Q, const _Float16* __restrict__ Kc,
    const _Float16* __restrict__ Vt, _Float16* __restrict__ G) {
  extern __shared__ char smem[];
  float*    S = reinterpret_cast<float*>(smem);
  _Float16* P = reinterpret_cast<_Float16*>(smem + S_BYTES);

  const int lane  = threadIdx.x, w = threadIdx.y;
  const int khalf = lane >> 4, lr = lane & 15;
  const int b  = blockIdx.x >> 5;
  const int q0 = (blockIdx.x & 31) << 5;
  const size_t base = (size_t)b << 20;   // b * 1024 * 1024
  const int n0 = w << 7;                 // 128-wide slab per wave

  // ---- phase 1: S = Q K^T * (1/32), 32 x 1024, wave slab of 128 keys ----
  {
    v8f acc[2][8];
#pragma unroll
    for (int i = 0; i < 2; ++i)
#pragma unroll
      for (int j = 0; j < 8; ++j) acc[i][j] = vzero8();
    const _Float16* Qp = Q  + base + (size_t)(q0 + lr) * 1024;
    const _Float16* Kp = Kc + base + (size_t)(n0 + lr) * 1024;
    for (int k0 = 0; k0 < 1024; k0 += 32) {
      v16h a0 = ldfrag(Qp + k0, khalf);
      v16h a1 = ldfrag(Qp + (size_t)16 * 1024 + k0, khalf);
#pragma unroll
      for (int j = 0; j < 8; ++j) {
        v16h bf = ldfrag(Kp + (size_t)(j << 4) * 1024 + k0, khalf);
        acc[0][j] = wmma_f16(a0, bf, acc[0][j]);
        acc[1][j] = wmma_f16(a1, bf, acc[1][j]);
      }
    }
    const float sc = 0.03125f;  // 1/sqrt(1024)
#pragma unroll
    for (int i = 0; i < 2; ++i)
#pragma unroll
      for (int j = 0; j < 8; ++j)
#pragma unroll
        for (int e = 0; e < 8; ++e) {
          int row = (i << 4) + (khalf << 3) + e;
          int key = n0 + (j << 4) + lr;
          S[(size_t)row * S_STRIDE + key] = acc[i][j][e] * sc;
        }
  }
  __syncthreads();

  // ---- phase 2: softmax ----
  softmax_rows(S, P, w, lane);
  __syncthreads();

  // ---- phase 3: O = P @ V  (wave slab: 128 output features) ----
  {
    v8f o[2][8];
#pragma unroll
    for (int i = 0; i < 2; ++i)
#pragma unroll
      for (int j = 0; j < 8; ++j) o[i][j] = vzero8();
    const _Float16* Vp = Vt + base + (size_t)(n0 + lr) * 1024;
    for (int k0 = 0; k0 < 1024; k0 += 32) {
      v16h a0 = ldfrag(P + (size_t)(lr)      * P_STRIDE + k0, khalf);
      v16h a1 = ldfrag(P + (size_t)(16 + lr) * P_STRIDE + k0, khalf);
#pragma unroll
      for (int j = 0; j < 8; ++j) {
        v16h bf = ldfrag(Vp + (size_t)(j << 4) * 1024 + k0, khalf);
        o[0][j] = wmma_f16(a0, bf, o[0][j]);
        o[1][j] = wmma_f16(a1, bf, o[1][j]);
      }
    }
#pragma unroll
    for (int i = 0; i < 2; ++i)
#pragma unroll
      for (int j = 0; j < 8; ++j)
#pragma unroll
        for (int e = 0; e < 8; ++e) {
          int row = q0 + (i << 4) + (khalf << 3) + e;
          int col = n0 + (j << 4) + lr;
          G[base + (size_t)row * 1024 + col] = (_Float16)o[i][j][e];
        }
  }
}

// ---------------------------------------------------------------------------
// LayerNorm over rows of 1024 (f16 in -> f16 out). One wave per row.
// ---------------------------------------------------------------------------
__global__ __launch_bounds__(256, 1) void layernorm_kernel(
    const _Float16* __restrict__ X, const float* __restrict__ g,
    const float* __restrict__ bta, _Float16* __restrict__ Y) {
  const int row = blockIdx.x * blockDim.y + threadIdx.y;
  const int lane = threadIdx.x;
  const _Float16* xr = X + (size_t)row * 1024;
  float x[32];
  float s = 0.f;
#pragma unroll
  for (int i = 0; i < 32; ++i) { x[i] = (float)xr[lane + (i << 5)]; s += x[i]; }
#pragma unroll
  for (int off = 16; off > 0; off >>= 1) s += __shfl_xor(s, off, 32);
  const float m = s * (1.f / 1024.f);
  float v = 0.f;
#pragma unroll
  for (int i = 0; i < 32; ++i) { float d = x[i] - m; v += d * d; }
#pragma unroll
  for (int off = 16; off > 0; off >>= 1) v += __shfl_xor(v, off, 32);
  const float rstd = rsqrtf(v * (1.f / 1024.f) + 1e-5f);
#pragma unroll
  for (int i = 0; i < 32; ++i) {
    int col = lane + (i << 5);
    Y[(size_t)row * 1024 + col] = (_Float16)((x[i] - m) * rstd * g[col] + bta[col]);
  }
}

// ---------------------------------------------------------------------------
// Multi-head attention: out = softmax(NQh NKh^T / 8) @ Vh, written directly in
// [B, SQ, H*DH] f32 layout.  One block per (b, h, 32-query tile).
// Phase 3 splits keys over the 8 waves; partials combined via ds_add_f32.
// ---------------------------------------------------------------------------
__global__ __launch_bounds__(256, 1) void mha_kernel(
    const _Float16* __restrict__ NQ, const _Float16* __restrict__ NK,
    const _Float16* __restrict__ Vt, float* __restrict__ Out) {
  extern __shared__ char smem[];
  float*    S = reinterpret_cast<float*>(smem);
  _Float16* P = reinterpret_cast<_Float16*>(smem + S_BYTES);
  float*    O = reinterpret_cast<float*>(smem + S_BYTES + P_BYTES);

  const int lane  = threadIdx.x, w = threadIdx.y;
  const int khalf = lane >> 4, lr = lane & 15;
  const int qt = blockIdx.x & 31;
  const int h  = (blockIdx.x >> 5) & 15;
  const int b  = blockIdx.x >> 9;
  const int q0 = qt << 5;
  const int rowbase = b * 1024;        // row offset into [8192,1024] activations
  const int n0 = w << 7;               // 128-key slab per wave (phase 1)
  const int tid = w * 32 + lane;

  // ---- phase 1: S = NQh NKh^T * (1/8) ----
  {
    v8f acc[2][8];
#pragma unroll
    for (int i = 0; i < 2; ++i)
#pragma unroll
      for (int j = 0; j < 8; ++j) acc[i][j] = vzero8();
    const _Float16* Qp = NQ + (size_t)(rowbase + q0 + lr) * 1024 + h * DH;
    const _Float16* Kp = NK + (size_t)(rowbase + n0 + lr) * 1024 + h * DH;
#pragma unroll
    for (int k0 = 0; k0 < DH; k0 += 32) {
      v16h a0 = ldfrag(Qp + k0, khalf);
      v16h a1 = ldfrag(Qp + (size_t)16 * 1024 + k0, khalf);
#pragma unroll
      for (int j = 0; j < 8; ++j) {
        v16h bf = ldfrag(Kp + (size_t)(j << 4) * 1024 + k0, khalf);
        acc[0][j] = wmma_f16(a0, bf, acc[0][j]);
        acc[1][j] = wmma_f16(a1, bf, acc[1][j]);
      }
    }
    const float sc = 0.125f;  // 1/sqrt(64)
#pragma unroll
    for (int i = 0; i < 2; ++i)
#pragma unroll
      for (int j = 0; j < 8; ++j)
#pragma unroll
        for (int e = 0; e < 8; ++e) {
          int row = (i << 4) + (khalf << 3) + e;
          int key = n0 + (j << 4) + lr;
          S[(size_t)row * S_STRIDE + key] = acc[i][j][e] * sc;
        }
  }
  // zero the output accumulator tile while S settles
  for (int idx = tid; idx < 32 * O_STRIDE; idx += 256) O[idx] = 0.f;
  __syncthreads();

  // ---- phase 2: softmax ----
  softmax_rows(S, P, w, lane);
  __syncthreads();

  // ---- phase 3: O += P[:, slab] @ Vh[slab, :]  (wave slab = 128 keys) ----
  {
    v8f o[2][4];
#pragma unroll
    for (int i = 0; i < 2; ++i)
#pragma unroll
      for (int j = 0; j < 4; ++j) o[i][j] = vzero8();
    const int kslab = w << 7;
    const _Float16* Vp =
        Vt + (size_t)(rowbase + h * DH + lr) * 1024;  // Vt row = feature n
#pragma unroll
    for (int ks = 0; ks < 4; ++ks) {
      const int k0 = kslab + (ks << 5);
      v16h a0 = ldfrag(P + (size_t)(lr)      * P_STRIDE + k0, khalf);
      v16h a1 = ldfrag(P + (size_t)(16 + lr) * P_STRIDE + k0, khalf);
#pragma unroll
      for (int j = 0; j < 4; ++j) {
        v16h bf = ldfrag(Vp + (size_t)(j << 4) * 1024 + k0, khalf);
        o[0][j] = wmma_f16(a0, bf, o[0][j]);
        o[1][j] = wmma_f16(a1, bf, o[1][j]);
      }
    }
#pragma unroll
    for (int i = 0; i < 2; ++i)
#pragma unroll
      for (int j = 0; j < 4; ++j)
#pragma unroll
        for (int e = 0; e < 8; ++e) {
          int row = (i << 4) + (khalf << 3) + e;
          int col = (j << 4) + lr;
          atomicAdd(&O[(size_t)row * O_STRIDE + col], o[i][j][e]);
        }
  }
  __syncthreads();

  for (int idx = tid; idx < 32 * DH; idx += 256) {
    int q = idx >> 6, dh = idx & 63;
    Out[(size_t)(rowbase + q0 + q) * 1024 + h * DH + dh] =
        O[(size_t)q * O_STRIDE + dh];
  }
}

// ---------------------------------------------------------------------------
// Host-side orchestration
// ---------------------------------------------------------------------------
extern "C" void kernel_launch(void* const* d_in, const int* in_sizes, int n_in,
                              void* d_out, int out_size, void* d_ws, size_t ws_size,
                              hipStream_t stream) {
  (void)in_sizes; (void)n_in; (void)out_size; (void)ws_size;

  const float* hidden  = (const float*)d_in[0];
  const float* context = (const float*)d_in[1];
  const float* Wq = (const float*)d_in[2];   const float* bq = (const float*)d_in[3];
  const float* Wk = (const float*)d_in[4];   const float* bk = (const float*)d_in[5];
  const float* Wv = (const float*)d_in[6];   const float* bv = (const float*)d_in[7];
  const float* Wobs = (const float*)d_in[8]; const float* bobs = (const float*)d_in[9];
  const float* Wmat = (const float*)d_in[10];const float* bmat = (const float*)d_in[11];
  const float* ln_g = (const float*)d_in[12];const float* ln_b = (const float*)d_in[13];
  float* out = (float*)d_out;

  char* ws = (char*)d_ws;
  const size_t MB = 1ull << 20;
  // Workspace layout (f16 unless noted), with aliasing of dead buffers:
  _Float16* WqH = (_Float16*)(ws + 0 * MB);
  _Float16* WkH = (_Float16*)(ws + 2 * MB);
  _Float16* WvH = (_Float16*)(ws + 4 * MB);
  _Float16* WoH = (_Float16*)(ws + 6 * MB);
  _Float16* WmH = (_Float16*)(ws + 8 * MB);
  _Float16* Hha = (_Float16*)(ws + 10 * MB);  // hidden f16; later aliased as T
  _Float16* Cha = (_Float16*)(ws + 26 * MB);  // context f16; later aliased as Vt
  _Float16* Qb  = (_Float16*)(ws + 42 * MB);  // q; later aliased as NQ
  _Float16* Kb  = (_Float16*)(ws + 58 * MB);  // k
  _Float16* Vb  = (_Float16*)(ws + 74 * MB);  // v
  _Float16* Gb  = (_Float16*)(ws + 90 * MB);  // guided ctx; later aliased as NK
  _Float16* Tb  = Hha;   // ReLU(GEMM) temp (pre-LN)
  _Float16* Vt  = Cha;   // V transposed per batch
  _Float16* NQ  = Qb;
  _Float16* NK  = Gb;

  const int NACT = Bsz * SQ * D;  // 8388608
  const int NW   = D * D;         // 1048576

  dim3 t256(256);
  // ---- 1. conversions ----
  f32_to_f16_kernel<<<NACT / 1024, t256, 0, stream>>>(hidden,  Hha, NACT);
  f32_to_f16_kernel<<<NACT / 1024, t256, 0, stream>>>(context, Cha, NACT);
  f32_to_f16_kernel<<<NW / 1024, t256, 0, stream>>>(Wq,   WqH, NW);
  f32_to_f16_kernel<<<NW / 1024, t256, 0, stream>>>(Wk,   WkH, NW);
  f32_to_f16_kernel<<<NW / 1024, t256, 0, stream>>>(Wv,   WvH, NW);
  f32_to_f16_kernel<<<NW / 1024, t256, 0, stream>>>(Wobs, WoH, NW);
  f32_to_f16_kernel<<<NW / 1024, t256, 0, stream>>>(Wmat, WmH, NW);

  // ---- 2. projections ----
  dim3 gblk(32, 8);
  const int M = Bsz * SQ;
  const int gemm_blocks = (M / 64) * (D / 64) / 8;  // 256
  gemm_bias_kernel<<<gemm_blocks, gblk, 0, stream>>>(Hha, WqH, bq, Qb, M, D, D, 0);
  gemm_bias_kernel<<<gemm_blocks, gblk, 0, stream>>>(Cha, WkH, bk, Kb, M, D, D, 0);
  gemm_bias_kernel<<<gemm_blocks, gblk, 0, stream>>>(Cha, WvH, bv, Vb, M, D, D, 0);

  // ---- 3. V transpose (per batch) ----
  transpose_kernel<<<dim3(16, 16, Bsz), dim3(64, 8), 0, stream>>>(Vb, Vt);

  // ---- 4. guide attention ----
  hipFuncSetAttribute(reinterpret_cast<const void*>(guide_attn_kernel),
                      hipFuncAttributeMaxDynamicSharedMemorySize, GUIDE_LDS);
  guide_attn_kernel<<<Bsz * (SQ / 32), gblk, GUIDE_LDS, stream>>>(Qb, Kb, Vt, Gb);

  // ---- 5. new_q = LN(ReLU(G @ Wobs^T + bobs)) ----
  gemm_bias_kernel<<<gemm_blocks, gblk, 0, stream>>>(Gb, WoH, bobs, Tb, M, D, D, 1);
  layernorm_kernel<<<M / 8, gblk, 0, stream>>>(Tb, ln_g, ln_b, NQ);

  // ---- 6. new_k = LN(ReLU(K @ Wmat^T + bmat)) ----
  gemm_bias_kernel<<<gemm_blocks, gblk, 0, stream>>>(Kb, WmH, bmat, Tb, M, D, D, 1);
  layernorm_kernel<<<M / 8, gblk, 0, stream>>>(Tb, ln_g, ln_b, NK);

  // ---- 7. multi-head attention -> out (f32) ----
  hipFuncSetAttribute(reinterpret_cast<const void*>(mha_kernel),
                      hipFuncAttributeMaxDynamicSharedMemorySize, MHA_LDS);
  mha_kernel<<<Bsz * Hh_ * (SQ / 32), gblk, MHA_LDS, stream>>>(NQ, NK, Vt, out);
}
